// Model_16930761081519
// MI455X (gfx1250) — compile-verified, matrix operand order checked by reading, not run on hardware
//
#include <hip/hip_runtime.h>
#include <cstdint>
#include <cstddef>

// ---------------------------------------------------------------------------
// Model dims (from reference): SEQ_LEN=64 B=128 EMB=512 H_ENC=512 NL=2 K=65536 D=512
// ---------------------------------------------------------------------------
#define TT   64
#define BB   128
#define EMBD 512
#define HH   512
#define DDIM 512
#define KCB  65536

typedef __attribute__((ext_vector_type(16))) __bf16 v16bf;
typedef __attribute__((ext_vector_type(8)))  __bf16 bf16x8;
typedef __attribute__((ext_vector_type(8)))  float  v8f;
typedef __attribute__((ext_vector_type(2)))  float  v2f;

// ----------------------------- helpers -------------------------------------
__device__ inline __bf16 f2bf(float f) {
  union { float f; uint32_t u; } v; v.f = f;
  uint32_t u = v.u;
  uint32_t r = (u + 0x7FFFu + ((u >> 16) & 1u)) >> 16;   // round-to-nearest-even
  union { unsigned short s; __bf16 b; } o; o.s = (unsigned short)r;
  return o.b;
}
__device__ inline float fast_sigmoid(float x) {
  x = fminf(15.f, fmaxf(-15.f, x));
  return 1.0f / (1.0f + __expf(-x));
}
__device__ inline float fast_tanh(float x) {
  x = fminf(15.f, fmaxf(-15.f, x));
  float e = __expf(2.0f * x);
  return (e - 1.0f) / (e + 1.0f);
}
// load one 16x32 bf16 WMMA fragment (per ISA 7.12.2: two 8-elem K-runs / lane)
__device__ inline v16bf ldfrag(const __bf16* p) {
  bf16x8 x0 = *(const bf16x8*)(p);
  bf16x8 x1 = *(const bf16x8*)(p + 16);
  v16bf r;
#pragma unroll
  for (int i = 0; i < 8; ++i) { r[i] = x0[i]; r[i + 8] = x1[i]; }
  return r;
}

// ----------------------------- converts / init -----------------------------
__global__ void k_f32_to_bf16(const float* __restrict__ src, __bf16* __restrict__ dst, int n) {
  int i = blockIdx.x * blockDim.x + threadIdx.x;
  if (i < n) dst[i] = f2bf(src[i]);
}
__global__ void k_zero_u32(uint32_t* p, int n) {
  int i = blockIdx.x * blockDim.x + threadIdx.x;
  if (i < n) p[i] = 0u;
}

// ---------------------------------------------------------------------------
// bf16 NT GEMM:  C[M,N] = init + A[M,K] * W[N,K]^T
//   init = Cinit[M,N] (f32, in-place allowed) or bias[N] or 0.
// One wave computes a 32x32 block = 2x2 WMMA tiles (register blocking halves
// L2 fragment traffic and gives 4 independent back-to-back v_wmma ops for
// XDL/VALU co-execution). All M,N used here are multiples of 32.
// Fragment layout per CDNA5 ISA 7.12.2: lane L: m(or n)=L&15, hi=L>>4;
// v16bf elems e<8 -> k=kt+8*hi+e, e>=8 -> k=kt+16+8*hi+(e-8).
// C/D: VGPR r holds (m = r + 8*hi, n = L&15).
// ---------------------------------------------------------------------------
__global__ void k_gemm_bf16_nt(const __bf16* __restrict__ A, int lda,
                               const __bf16* __restrict__ W, int ldb,
                               const float* __restrict__ bias,
                               const float* Cinit, int ldci,
                               float* C, int ldc,
                               int M, int N, int K)
{
  const int lane = threadIdx.x & 31;
  const int wave = threadIdx.x >> 5;                 // 8 waves / block
  const int bN = N >> 5;
  const int bM = M >> 5;
  const int blk = blockIdx.x * 8 + wave;
  if (blk >= bM * bN) return;                        // wave-uniform: EXEC stays full
  const int mb = blk / bN;
  const int nb = blk - mb * bN;
  const int hi  = lane >> 4;
  const int l15 = lane & 15;

  const __bf16* pa0 = A + (size_t)(mb * 32 + l15) * lda;
  const __bf16* pa1 = pa0 + (size_t)16 * lda;
  const __bf16* pw0 = W + (size_t)(nb * 32 + l15) * ldb;
  const __bf16* pw1 = pw0 + (size_t)16 * ldb;
  const int cn0 = nb * 32 + l15, cn1 = cn0 + 16;

  v8f acc00, acc01, acc10, acc11;
#pragma unroll
  for (int r = 0; r < 8; ++r) {
    int cm0 = mb * 32 + r + hi * 8, cm1 = cm0 + 16;
    if (Cinit) {
      acc00[r] = Cinit[(size_t)cm0 * ldci + cn0];
      acc01[r] = Cinit[(size_t)cm0 * ldci + cn1];
      acc10[r] = Cinit[(size_t)cm1 * ldci + cn0];
      acc11[r] = Cinit[(size_t)cm1 * ldci + cn1];
    } else if (bias) {
      float b0 = bias[cn0], b1 = bias[cn1];
      acc00[r] = b0; acc01[r] = b1; acc10[r] = b0; acc11[r] = b1;
    } else {
      acc00[r] = 0.f; acc01[r] = 0.f; acc10[r] = 0.f; acc11[r] = 0.f;
    }
  }

  for (int kt = 0; kt < K; kt += 32) {
    const int k0 = kt + hi * 8;
    v16bf a0 = ldfrag(pa0 + k0);
    v16bf a1 = ldfrag(pa1 + k0);
    v16bf b0 = ldfrag(pw0 + k0);
    v16bf b1 = ldfrag(pw1 + k0);
    __builtin_prefetch(pw0 + k0 + 64, 0, 1);         // global_prefetch_b8: K tile two ahead
    acc00 = __builtin_amdgcn_wmma_f32_16x16x32_bf16(false, a0, false, b0, (short)0, acc00, false, false);
    acc01 = __builtin_amdgcn_wmma_f32_16x16x32_bf16(false, a0, false, b1, (short)0, acc01, false, false);
    acc10 = __builtin_amdgcn_wmma_f32_16x16x32_bf16(false, a1, false, b0, (short)0, acc10, false, false);
    acc11 = __builtin_amdgcn_wmma_f32_16x16x32_bf16(false, a1, false, b1, (short)0, acc11, false, false);
  }

#pragma unroll
  for (int r = 0; r < 8; ++r) {
    int cm0 = mb * 32 + r + hi * 8, cm1 = cm0 + 16;
    C[(size_t)cm0 * ldc + cn0] = acc00[r];
    C[(size_t)cm0 * ldc + cn1] = acc01[r];
    C[(size_t)cm1 * ldc + cn0] = acc10[r];
    C[(size_t)cm1 * ldc + cn1] = acc11[r];
  }
}

// ---------------------------------------------------------------------------
// LSTM pointwise cell: gates G[B,4H] in torch order (i,f,g,o).
//   c = sig(f)*c + sig(i)*tanh(g);  h = sig(o)*tanh(c)
// ---------------------------------------------------------------------------
__global__ void k_lstm_cell(const float* __restrict__ G,
                            float* __restrict__ c, float* __restrict__ h,
                            __bf16* __restrict__ hbf,
                            __bf16* seq, int ldseq, int H)
{
  int i = blockIdx.x * blockDim.x + threadIdx.x;     // exactly B*H threads
  int b = i / H, j = i - b * H;
  const float* g = G + (size_t)b * 4 * H;
  float gi = g[j], gf = g[H + j], gg = g[2 * H + j], go = g[3 * H + j];
  float cn = fast_sigmoid(gf) * c[i] + fast_sigmoid(gi) * fast_tanh(gg);
  float hn = fast_sigmoid(go) * fast_tanh(cn);
  c[i] = cn; h[i] = hn;
  __bf16 hb = f2bf(hn);
  hbf[i] = hb;
  if (seq) seq[(size_t)b * ldseq + j] = hb;
}

// concat [hA|hB|hC|hD] (each [B,H] f32) -> bf16 [B,4H]
__global__ void k_concat4_bf16(const float* __restrict__ a, const float* __restrict__ b,
                               const float* __restrict__ c, const float* __restrict__ d,
                               __bf16* __restrict__ out, int H)
{
  int i = blockIdx.x * blockDim.x + threadIdx.x;     // B*4H threads
  int W4 = 4 * H;
  int row = i / W4, col = i - row * W4;
  const float* s = (col < H) ? a : (col < 2 * H) ? b : (col < 3 * H) ? c : d;
  out[i] = f2bf(s[(size_t)row * H + (col & (H - 1))]);
}

// 0.5*||cb_row||^2 per code; one wave per code (coalesced strided lanes)
__global__ void k_halfnorm(const float* __restrict__ cb, float* __restrict__ out, int K, int Dd)
{
  int lane = threadIdx.x & 31;
  int code = blockIdx.x * (blockDim.x >> 5) + (threadIdx.x >> 5);
  if (code >= K) return;
  const float* r = cb + (size_t)code * Dd;
  float s = 0.f;
  for (int j = lane; j < Dd; j += 32) { float v = r[j]; s += v * v; }
#pragma unroll
  for (int m = 16; m; m >>= 1) s += __shfl_xor(s, m, 32);
  if (lane == 0) out[code] = 0.5f * s;
}

// ---------------------------------------------------------------------------
// Fused VQ scoring + per-row chunk argmax.
// score[m,n] = z[m]·cb[n] - 0.5||cb[n]||^2  (argmax == argmin of L2 distance)
// One block handles all 128 rows x a 16-code chunk.
// The block's 32KB codebook tile (16 contiguous rows) is staged into LDS once
// via the CDNA5 async path (global_load_async_to_lds_b128 + s_wait_asynccnt),
// removing the 8x-redundant per-wave global reads of the bandwidth-bound
// codebook stream. B fragments then come from LDS (ds_load_b64).
// Uses v_wmma_f32_16x16x4_f32 (f32 precision keeps argmin stable).
// ---------------------------------------------------------------------------
__global__ void k_vq_score(const float* __restrict__ Z, const float* __restrict__ CB,
                           const float* __restrict__ halfnorm,
                           float* __restrict__ pval, int* __restrict__ pidx,
                           int Kd, int nChunks)
{
  __shared__ __align__(16) float cbT[16 * 512];      // 32KB codebook tile
  __shared__ float sTile[128][17];
  const int tid  = threadIdx.x;
  const int lane = tid & 31;
  const int wave = tid >> 5;                         // 8 waves -> rows 0..127
  const int hi = lane >> 4, l15 = lane & 15;
  const int nbase = blockIdx.x * 16;

  // ---- async stage: 16 rows x 512 f32 = 2048 x 16B chunks, 8 per thread ----
  {
    const float* gsrc = CB + (size_t)nbase * Kd;     // 16 contiguous rows
    uint32_t ldsBase = (uint32_t)(uintptr_t)(&cbT[0]);
#pragma unroll
    for (int i = 0; i < 8; ++i) {
      int chunk = tid + i * 256;
      uint32_t lo = ldsBase + (uint32_t)chunk * 16u;
      uint64_t ga = (uint64_t)(uintptr_t)gsrc + (uint64_t)chunk * 16u;
      asm volatile("global_load_async_to_lds_b128 %0, %1, off"
                   :: "v"(lo), "v"(ga) : "memory");
    }
    asm volatile("s_wait_asynccnt 0x0" ::: "memory");
  }
  __syncthreads();

  const float* pz = Z + (size_t)(wave * 16 + l15) * Kd;
  const float* pb = &cbT[(size_t)l15 * Kd];
  v8f acc = {};
  for (int kt = 0; kt < Kd; kt += 4) {
    v2f a = *(const v2f*)(pz + kt + hi * 2);
    v2f b = *(const v2f*)(pb + kt + hi * 2);
    acc = __builtin_amdgcn_wmma_f32_16x16x4_f32(false, a, false, b, (short)0, acc, false, false);
  }
  float hn = halfnorm[nbase + l15];
#pragma unroll
  for (int r = 0; r < 8; ++r) sTile[wave * 16 + r + hi * 8][l15] = acc[r] - hn;
  __syncthreads();
  if (tid < 128) {
    int row = tid;
    float best = sTile[row][0]; int bi = 0;
#pragma unroll
    for (int n = 1; n < 16; ++n) { float v = sTile[row][n]; if (v > best) { best = v; bi = n; } }
    pval[(size_t)row * nChunks + blockIdx.x] = best;
    pidx[(size_t)row * nChunks + blockIdx.x] = nbase + bi;
  }
}

__global__ void k_vq_reduce(const float* __restrict__ pval, const int* __restrict__ pidx,
                            int nChunks, int* __restrict__ bestIdx)
{
  __shared__ float sv[256]; __shared__ int si[256];
  int row = blockIdx.x, t = threadIdx.x;
  float best = -3.4e38f; int bi = 0x7fffffff;
  for (int c = t; c < nChunks; c += 256) {
    float v = pval[(size_t)row * nChunks + c];
    int  ix = pidx[(size_t)row * nChunks + c];
    if (v > best || (v == best && ix < bi)) { best = v; bi = ix; }
  }
  sv[t] = best; si[t] = bi; __syncthreads();
  for (int s = 128; s > 0; s >>= 1) {
    if (t < s && (sv[t + s] > sv[t] || (sv[t + s] == sv[t] && si[t + s] < si[t]))) { sv[t] = sv[t + s]; si[t] = si[t + s]; }
    __syncthreads();
  }
  if (t == 0) bestIdx[row] = si[0];
}

// zq = cb[idx]; write to d_out region + decoder state (f32 and optional bf16)
__global__ void k_vq_gather(const float* __restrict__ cb, const int* __restrict__ bestIdx,
                            float* __restrict__ zq, float* __restrict__ state,
                            __bf16* statebf, int Dd)
{
  int i = blockIdx.x * blockDim.x + threadIdx.x;     // B*D threads
  int row = i / Dd, j = i - row * Dd;
  float v = cb[(size_t)bestIdx[row] * Dd + j];
  zq[i] = v; state[i] = v;
  if (statebf) statebf[i] = f2bf(v);
}

// ----------------------------- host-side glue ------------------------------
static void gemm(hipStream_t s, const __bf16* A, int lda, const __bf16* W, int ldb,
                 const float* bias, const float* Cinit, int ldci,
                 float* C, int ldc, int M, int N, int K)
{
  int tiles = (M >> 5) * (N >> 5);                   // 32x32 per wave
  k_gemm_bf16_nt<<<(tiles + 7) / 8, 256, 0, s>>>(A, lda, W, ldb, bias, Cinit, ldci, C, ldc, M, N, K);
}

extern "C" void kernel_launch(void* const* d_in, const int* in_sizes, int n_in,
                              void* d_out, int out_size, void* d_ws, size_t ws_size,
                              hipStream_t stream)
{
  (void)in_sizes; (void)n_in; (void)out_size; (void)ws_size;

  // Input order = setup_inputs() dict order, nested tuples flattened:
  // x, [enc L0 fwd (Wih,Whh,b), L0 bwd, L1 fwd, L1 bwd], fc_W, fc_b, cb_h, cb_c,
  // dec (Wih,Whh,b), out_W, out_b
  const float* x = (const float*)d_in[0];
  const float* eW[2][2][3];
  { int p = 1;
    for (int l = 0; l < 2; ++l) for (int d = 0; d < 2; ++d) for (int q = 0; q < 3; ++q)
      eW[l][d][q] = (const float*)d_in[p++]; }
  const float* fcW  = (const float*)d_in[13];
  const float* fcb  = (const float*)d_in[14];
  const float* cbh  = (const float*)d_in[15];
  const float* cbc  = (const float*)d_in[16];
  const float* dWih = (const float*)d_in[17];
  const float* dWhh = (const float*)d_in[18];
  const float* dbias= (const float*)d_in[19];
  const float* oW   = (const float*)d_in[20];
  const float* ob   = (const float*)d_in[21];

  float* dout = (float*)d_out;
  float* zeh = dout + (size_t)TT * BB * EMBD;        // outputs is [64,128,512]
  float* zec = zeh + BB * DDIM;
  float* zqh = zec + BB * DDIM;
  float* zqc = zqh + BB * DDIM;

  // ---- workspace carve-out ----
  uint8_t* wsb = (uint8_t*)d_ws; size_t off = 0;
  auto alloc = [&](size_t bytes) -> void* {
    void* p = wsb + off; off = (off + bytes + 255) & ~(size_t)255; return p;
  };
  __bf16* xbf = (__bf16*)alloc((size_t)TT * BB * EMBD * 2);
  __bf16 *encWihBf[2][2], *encWhhBf[2][2];
  for (int l = 0; l < 2; ++l) for (int d = 0; d < 2; ++d) {
    int indim = (l == 0) ? EMBD : 2 * HH;
    encWihBf[l][d] = (__bf16*)alloc((size_t)4 * HH * indim * 2);
    encWhhBf[l][d] = (__bf16*)alloc((size_t)4 * HH * HH * 2);
  }
  __bf16* decWihBf = (__bf16*)alloc((size_t)4 * DDIM * EMBD * 2);
  __bf16* decWhhBf = (__bf16*)alloc((size_t)4 * DDIM * DDIM * 2);
  __bf16* fcWbf    = (__bf16*)alloc((size_t)DDIM * 4 * HH * 2);
  __bf16* oWbf     = (__bf16*)alloc((size_t)EMBD * DDIM * 2);
  __bf16* seq      = (__bf16*)alloc((size_t)TT * BB * 2 * HH * 2);   // L0 out / L1 in
  float*  gatesF   = (float*)alloc((size_t)TT * BB * 4 * HH * 4);
  float*  gatesB   = (float*)alloc((size_t)TT * BB * 4 * HH * 4);
  const size_t stStride = (size_t)BB * HH * 4 * 2 + (size_t)BB * HH * 2; // h,c f32 + h bf16
  uint8_t* stBase = (uint8_t*)alloc(stStride * 5);
  float* hS[5]; float* cS[5]; __bf16* hbS[5];
  for (int s = 0; s < 5; ++s) {
    hS[s]  = (float*)(stBase + s * stStride);
    cS[s]  = hS[s] + BB * HH;
    hbS[s] = (__bf16*)(cS[s] + BB * HH);
  }
  __bf16* catH = (__bf16*)alloc((size_t)BB * 4 * HH * 2);
  __bf16* catC = (__bf16*)alloc((size_t)BB * 4 * HH * 2);
  float* hnH = (float*)alloc((size_t)KCB * 4);
  float* hnC = (float*)alloc((size_t)KCB * 4);
  const int nChunks = KCB / 16;
  float* pval = (float*)alloc((size_t)BB * nChunks * 4);
  int*   pidx = (int*)alloc((size_t)BB * nChunks * 4);
  int*   bidx = (int*)alloc(BB * 4);
  __bf16* ysbf = (__bf16*)alloc((size_t)(TT - 1) * BB * DDIM * 2);

  // ---- bf16 conversions (weights + x) ----
  auto conv = [&](const float* s, __bf16* d, size_t n) {
    k_f32_to_bf16<<<(int)((n + 255) / 256), 256, 0, stream>>>(s, d, (int)n);
  };
  conv(x, xbf, (size_t)TT * BB * EMBD);
  for (int l = 0; l < 2; ++l) for (int d = 0; d < 2; ++d) {
    int indim = (l == 0) ? EMBD : 2 * HH;
    conv(eW[l][d][0], encWihBf[l][d], (size_t)4 * HH * indim);
    conv(eW[l][d][1], encWhhBf[l][d], (size_t)4 * HH * HH);
  }
  conv(dWih, decWihBf, (size_t)4 * DDIM * EMBD);
  conv(dWhh, decWhhBf, (size_t)4 * DDIM * DDIM);
  conv(fcW,  fcWbf,    (size_t)DDIM * 4 * HH);
  conv(oW,   oWbf,     (size_t)EMBD * DDIM);

  // zero the 4 encoder (h,c,h_bf16) state sets (must re-zero every call)
  { int n = (int)(4 * stStride / 4);
    k_zero_u32<<<(n + 255) / 256, 256, 0, stream>>>((uint32_t*)stBase, n); }

  // ---- encoder: 2 bidirectional layers ----
  const int cellBlocks = (BB * HH) / 256;
  for (int l = 0; l < 2; ++l) {
    const __bf16* inSeq = (l == 0) ? xbf : seq;
    const int inDim = (l == 0) ? EMBD : 2 * HH;
    // batched input-side gates for both directions: [T*B, 4H]
    gemm(stream, inSeq, inDim, encWihBf[l][0], inDim, eW[l][0][2], nullptr, 0,
         gatesF, 4 * HH, TT * BB, 4 * HH, inDim);
    gemm(stream, inSeq, inDim, encWihBf[l][1], inDim, eW[l][1][2], nullptr, 0,
         gatesB, 4 * HH, TT * BB, 4 * HH, inDim);
    const int sf = l * 2, sb = l * 2 + 1;
    for (int t = 0; t < TT; ++t) {
      // forward direction, time t
      float* GF = gatesF + (size_t)t * BB * 4 * HH;
      gemm(stream, hbS[sf], HH, encWhhBf[l][0], HH, nullptr, GF, 4 * HH,
           GF, 4 * HH, BB, 4 * HH, HH);
      k_lstm_cell<<<cellBlocks, 256, 0, stream>>>(GF, cS[sf], hS[sf], hbS[sf],
          (l == 0) ? seq + (size_t)t * BB * 2 * HH : nullptr, 2 * HH, HH);
      // backward direction processes original time index T-1-t
      int tb = TT - 1 - t;
      float* GB = gatesB + (size_t)tb * BB * 4 * HH;
      gemm(stream, hbS[sb], HH, encWhhBf[l][1], HH, nullptr, GB, 4 * HH,
           GB, 4 * HH, BB, 4 * HH, HH);
      k_lstm_cell<<<cellBlocks, 256, 0, stream>>>(GB, cS[sb], hS[sb], hbS[sb],
          (l == 0) ? seq + (size_t)tb * BB * 2 * HH + HH : nullptr, 2 * HH, HH);
    }
  }

  // ---- FC bottleneck: z_e = cat(h states)[B,2048] @ fc_W^T + fc_b ----
  { int n = BB * 4 * HH;
    k_concat4_bf16<<<n / 256, 256, 0, stream>>>(hS[0], hS[1], hS[2], hS[3], catH, HH);
    k_concat4_bf16<<<n / 256, 256, 0, stream>>>(cS[0], cS[1], cS[2], cS[3], catC, HH); }
  gemm(stream, catH, 4 * HH, fcWbf, 4 * HH, fcb, nullptr, 0, zeh, DDIM, BB, DDIM, 4 * HH);
  gemm(stream, catC, 4 * HH, fcWbf, 4 * HH, fcb, nullptr, 0, zec, DDIM, BB, DDIM, 4 * HH);

  // ---- VQ: fused score+argmax (f32 WMMA, async-LDS codebook staging) ----
  k_halfnorm<<<KCB / 8, 256, 0, stream>>>(cbh, hnH, KCB, DDIM);
  k_halfnorm<<<KCB / 8, 256, 0, stream>>>(cbc, hnC, KCB, DDIM);
  k_vq_score<<<nChunks, 256, 0, stream>>>(zeh, cbh, hnH, pval, pidx, DDIM, nChunks);
  k_vq_reduce<<<BB, 256, 0, stream>>>(pval, pidx, nChunks, bidx);
  k_vq_gather<<<(BB * DDIM) / 256, 256, 0, stream>>>(cbh, bidx, zqh, hS[4], hbS[4], DDIM);
  k_vq_score<<<nChunks, 256, 0, stream>>>(zec, cbc, hnC, pval, pidx, DDIM, nChunks);
  k_vq_reduce<<<BB, 256, 0, stream>>>(pval, pidx, nChunks, bidx);
  k_vq_gather<<<(BB * DDIM) / 256, 256, 0, stream>>>(cbc, bidx, zqc, cS[4], nullptr, DDIM);

  // ---- decoder LSTM over x[:-1], init (h,c) = (zq_h, zq_c) ----
  gemm(stream, xbf, EMBD, decWihBf, EMBD, dbias, nullptr, 0,
       gatesF, 4 * DDIM, (TT - 1) * BB, 4 * DDIM, EMBD);
  for (int t = 0; t < TT - 1; ++t) {
    float* G = gatesF + (size_t)t * BB * 4 * DDIM;
    gemm(stream, hbS[4], DDIM, decWhhBf, DDIM, nullptr, G, 4 * DDIM,
         G, 4 * DDIM, BB, 4 * DDIM, DDIM);
    k_lstm_cell<<<cellBlocks, 256, 0, stream>>>(G, cS[4], hS[4], hbS[4],
        ysbf + (size_t)t * BB * DDIM, DDIM, DDIM);
  }

  // ---- output projection: outs = ys @ out_W^T + out_b -> outputs[1:] ----
  gemm(stream, ysbf, DDIM, oWbf, DDIM, ob, nullptr, 0,
       dout + (size_t)BB * EMBD, EMBD, (TT - 1) * BB, EMBD, DDIM);

  // outputs[0] = x[0]
  hipMemcpyAsync(dout, x, (size_t)BB * EMBD * sizeof(float),
                 hipMemcpyDeviceToDevice, stream);
}
